// SNNDirectionDecoder_11854109737296
// MI455X (gfx1250) — matrix operand order repaired and checked by reading, not compiled, not run on hardware
//
#include <hip/hip_runtime.h>

// CDNA5 / gfx1250 fused SNN decoder, round 2:
//  - cur1 = x @ W1^T + b1 per time step via v_wmma_f32_16x16x32_bf16
//  - x tiles double-buffered in LDS, staged by the async global->LDS engine
//    (global_load_async_to_lds_b128 + s_wait_asynccnt), overlapping HBM
//    latency with WMMA compute across the T=500 recurrence.
//  - LIF recurrence state lives in the WMMA C/D VGPR layout.
//  - out = spk_sum @ W2^T + T*b2 via LDS reduce + global f32 atomics.

typedef __attribute__((ext_vector_type(16))) __bf16 v16bf;
typedef __attribute__((ext_vector_type(8)))  __bf16 v8bf;
typedef __attribute__((ext_vector_type(8)))  float  v8f;

#define B_   256
#define T_   500
#define I_   256
#define H_   512
#define C_   8
#define HCOLS 64          // H columns per workgroup (16 per wave)
#define NTHREADS 128      // 4 waves (wave32)

__device__ __forceinline__ void async_load_b128_to_lds(const void* gaddr,
                                                       unsigned lds_addr) {
  // VGLOBAL async DMA: copies 16 B from global vaddr into LDS at vdst, tracked by ASYNCcnt.
  asm volatile("global_load_async_to_lds_b128 %0, %1, off"
               :: "v"(lds_addr), "v"(gaddr) : "memory");
}

__device__ __forceinline__ void wait_asynccnt0() {
  asm volatile("s_wait_asynccnt 0" ::: "memory");
}

__global__ void __launch_bounds__(NTHREADS) snn_zero_kernel(float* p, int n) {
  int i = blockIdx.x * blockDim.x + threadIdx.x;
  if (i < n) p[i] = 0.0f;
}

__global__ void __launch_bounds__(NTHREADS) snn_fused_kernel(
    const float* __restrict__ x,   // (B, T, I)
    const float* __restrict__ W1,  // (H, I)
    const float* __restrict__ b1,  // (H)
    const float* __restrict__ W2,  // (C, H)
    const float* __restrict__ b2,  // (C)
    float* __restrict__ out)       // (B, C), pre-zeroed
{
  __shared__ __bf16 w1s[HCOLS * I_];   // 32 KB, resident W1 slice (bf16)
  __shared__ float  xf[2][16 * I_];    // 2 x 16 KB f32 x-tiles (double buffer)
  float* sums = xf[0];                 // reused post-loop: 16 x HCOLS f32

  const int tid   = threadIdx.x;
  const int wave  = tid >> 5;
  const int lane  = tid & 31;
  const int half  = lane >> 4;   // 0: lanes 0-15, 1: lanes 16-31
  const int mrow  = lane & 15;   // A-row (M) / C-col (N)
  const int bbase = blockIdx.x * 16;
  const int hbase = blockIdx.y * HCOLS;

  // Stage this block's W1 slice into LDS as bf16, [col][k] layout.
  for (int i = tid; i < HCOLS * I_; i += NTHREADS) {
    const int c = i >> 8;          // I_ == 256
    const int k = i & (I_ - 1);
    w1s[i] = (__bf16)W1[(size_t)(hbase + c) * I_ + k];
  }

  // LDS byte addresses of the two x-tile buffers (wave-relative, low 32 bits).
  const unsigned xlds0 = (unsigned)(size_t)&xf[0][0];
  const unsigned xlds1 = (unsigned)(size_t)&xf[1][0];

  // Issue async DMA of one 16x256 f32 tile (16 KB): 8 x b128 per thread.
  auto stage_async = [&](int t, unsigned lds_base) {
#pragma unroll
    for (int j = 0; j < 8; ++j) {
      const int o  = (tid + j * NTHREADS) * 16;  // byte offset within tile
      const int m  = o >> 10;                    // 1024 B per tile row
      const int kb = o & 1023;                   // byte within row
      const char* g = (const char*)x +
          (((size_t)(bbase + m) * (T_ * I_) + (size_t)t * I_) << 2) + kb;
      async_load_b128_to_lds(g, lds_base + (unsigned)o);
    }
  };

  // Prologue: DMA tile 0, then make it (and w1s) visible to all waves.
  stage_async(0, xlds0);
  wait_asynccnt0();
  __syncthreads();

  const int   col = wave * 16 + mrow;   // this lane's local H column
  const float b1v = b1[hbase + col];

  v8f mem  = {};   // membrane potential, (M = r + 8*half, N = col)
  v8f ssum = {};   // spike count accumulator

  int cur = 0;
  for (int t = 0; t < T_; ++t) {
    const int nxt = cur ^ 1;
    // Kick off DMA of tile t+1 into the other buffer (overlaps WMMA below).
    if (t + 1 < T_) stage_async(t + 1, nxt ? xlds1 : xlds0);
    // Pull tile t+2 toward L2 ahead of the async engine.
    if (t + 2 < T_) {
      const int o = tid * 128;        // 128 threads x 128 B = 16 KB
      const int m = o >> 10, kb = o & 1023;
      __builtin_prefetch((const char*)x +
          (((size_t)(bbase + m) * (T_ * I_) + (size_t)(t + 2) * I_) << 2) + kb,
          0, 0);
    }

    const float* xt = xf[cur];
    v8f acc = {};
#pragma unroll
    for (int kk = 0; kk < I_ / 32; ++kk) {
      const int kc = kk * 32;
      // A fragment (16x32): two 32B f32 LDS loads, packed-convert to bf16.
      const v8f alo = *(const v8f*)&xt[mrow * I_ + kc + 8 * half];
      const v8f ahi = *(const v8f*)&xt[mrow * I_ + kc + 16 + 8 * half];
      v16bf a;
#pragma unroll
      for (int e = 0; e < 8; ++e) {
        a[e]     = (__bf16)alo[e];   // VGPRs 0-3: localK 8*half..+8
        a[8 + e] = (__bf16)ahi[e];   // VGPRs 4-7: localK 16+8*half..+8
      }
      // B fragment (32x16 bf16): two 16B LDS loads per lane, localK 8*half..+8 and 16+8*half..+8.
      const v8bf blo = *(const v8bf*)&w1s[col * I_ + kc + 8 * half];
      const v8bf bhi = *(const v8bf*)&w1s[col * I_ + kc + 16 + 8 * half];
      const v16bf bf = __builtin_shufflevector(blo, bhi, 0, 1, 2, 3, 4, 5, 6, 7, 8, 9, 10, 11, 12, 13, 14, 15);
      acc = __builtin_amdgcn_wmma_f32_16x16x32_bf16(
          /*neg_a=*/false, a, /*neg_b=*/false, bf,
          /*c_mod=*/(short)0, acc, /*reuse_a=*/false, /*reuse_b=*/false);
    }

    // Leaky-integrate-fire (beta=0.9, threshold=1, reset='subtract'):
    //   reset = H(mem - 1); mem = 0.9*mem + cur - reset; spk = H(mem - 1)
#pragma unroll
    for (int r = 0; r < 8; ++r) {
      const float curv  = acc[r] + b1v;
      const float m0    = mem[r];
      const float reset = (m0 > 1.0f) ? 1.0f : 0.0f;
      const float mn    = 0.9f * m0 + curv - reset;
      mem[r]  = mn;
      ssum[r] += (mn > 1.0f) ? 1.0f : 0.0f;
    }

    // Our async writes done, then all waves' writes visible -> safe to swap.
    wait_asynccnt0();
    __syncthreads();
    cur = nxt;
  }

  // Dump spike sums (16 rows x HCOLS cols) to LDS.
#pragma unroll
  for (int r = 0; r < 8; ++r) {
    const int M = r + 8 * half;
    sums[M * HCOLS + col] = ssum[r];
  }
  __syncthreads();

  // out[b, c] += sum_j spk_sum[b, j] * W2[c, hbase + j]  (+ T*b2 once).
  {
    const int row = tid >> 3;   // 0..15
    const int c   = tid & 7;    // 0..7
    float v = 0.0f;
#pragma unroll 4
    for (int j = 0; j < HCOLS; ++j)
      v += sums[row * HCOLS + j] * W2[c * H_ + hbase + j];
    if (blockIdx.y == 0) v += (float)T_ * b2[c];
    atomicAdd(&out[(bbase + row) * C_ + c], v);
  }
}

extern "C" void kernel_launch(void* const* d_in, const int* in_sizes, int n_in,
                              void* d_out, int out_size, void* d_ws, size_t ws_size,
                              hipStream_t stream) {
  const float* x  = (const float*)d_in[0];
  const float* W1 = (const float*)d_in[1];
  const float* b1 = (const float*)d_in[2];
  const float* W2 = (const float*)d_in[3];
  const float* b2 = (const float*)d_in[4];
  float* out = (float*)d_out;

  // d_out is poisoned; atomics below require zero-init first (same stream).
  snn_zero_kernel<<<(out_size + NTHREADS - 1) / NTHREADS, NTHREADS, 0, stream>>>(
      out, out_size);

  dim3 grid(B_ / 16, H_ / HCOLS);   // 16 x 8 = 128 workgroups
  snn_fused_kernel<<<grid, NTHREADS, 0, stream>>>(x, W1, b1, W2, b2, out);
}
